// BoxHead_49048526521000
// MI455X (gfx1250) — matrix operand order repaired and checked by reading, not compiled, not run on hardware
//
#include <hip/hip_runtime.h>
#include <hip/hip_bf16.h>
#include <math.h>

typedef __attribute__((ext_vector_type(16))) __bf16 v16bf;
typedef __attribute__((ext_vector_type(8)))  __bf16 v8bf;
typedef __attribute__((ext_vector_type(4)))  __bf16 v4bf;
typedef __attribute__((ext_vector_type(8)))  float  v8f;
typedef __attribute__((ext_vector_type(4)))  float  v4f;

#define M_BLK 128
#define N_BLK 128
#define K_BLK 32
#define KPAD  40   // LDS row stride in bf16 elems: 80 B = 20 banks; gcd(20,64)=4
                   // -> 16 consecutive rows hit 16 distinct banks; 16B-aligned.

static __device__ inline v4bf cvt4(const v4f v) {
    return v4bf{(__bf16)v.x, (__bf16)v.y, (__bf16)v.z, (__bf16)v.w};
}

// C[M,N] = op(A[M,K] @ B[K,N] + bias[N]), op = identity or ReLU.
// fp32 in/out; A,B converted to bf16 on the fly; f32 accumulation via
// v_wmma_f32_16x16x32_bf16.  Block tile 128x128, 8 waves of 32; each wave
// owns a 64x32 sub-tile = 4x2 WMMA accumulators.  Double-buffered LDS,
// software-pipelined: global loads of tile t+1 overlap WMMA on tile t,
// single barrier per stage.
__global__ __launch_bounds__(256) void gemm_bias_kernel(
    const float* __restrict__ A, const float* __restrict__ B,
    const float* __restrict__ bias, float* __restrict__ C,
    int M, int N, int K, int relu)
{
    __shared__ __align__(16) __bf16 lds_a[2][M_BLK * KPAD];  // [m][k] row-major
    __shared__ __align__(16) __bf16 lds_b[2][N_BLK * KPAD];  // [n][k] transposed

    const int tid  = threadIdx.x;
    const int row0 = blockIdx.y * M_BLK;
    const int col0 = blockIdx.x * N_BLK;

    const int wave = tid >> 5;
    const int lane = tid & 31;
    const int wm   = wave & 1;                  // 2 wave-rows of 64
    const int wn   = wave >> 1;                 // 4 wave-cols of 32
    const int l16  = lane & 15;
    const int kba  = (lane < 16) ? 0 : 8;       // A fragment k-base (ISA layout)
    const int khb  = (lane < 16) ? 0 : 16;      // unused; B fragment also uses kba (same k order as A)

    // Per-thread staging indices (loop-invariant).
    int rA[4], cA[4], nB[4], kqB[4];
    #pragma unroll
    for (int i = 0; i < 4; ++i) {
        const int ia = tid + i * 256;           // 0..1023
        rA[i]  = ia >> 3;                       // 0..127
        cA[i]  = (ia & 7) << 2;                 // 0,4,..,28
        nB[i]  = ia & 127;                      // 0..127
        kqB[i] = ia >> 7;                       // 0..7  (k quad)
    }

    v8f acc[4][2] = {};

    const int nk = K / K_BLK;

    // ---- prologue: stage tile 0 into buffer 0 ----
    {
        #pragma unroll
        for (int i = 0; i < 4; ++i) {
            const v4f a4 = *(const v4f*)(A + (size_t)(row0 + rA[i]) * K + cA[i]);
            *(v4bf*)&lds_a[0][rA[i] * KPAD + cA[i]] = cvt4(a4);
        }
        #pragma unroll
        for (int i = 0; i < 4; ++i) {
            v4f b4;
            #pragma unroll
            for (int j = 0; j < 4; ++j)
                b4[j] = B[(size_t)(kqB[i] * 4 + j) * N + col0 + nB[i]];
            *(v4bf*)&lds_b[0][nB[i] * KPAD + kqB[i] * 4] = cvt4(b4);
        }
    }
    __syncthreads();

    int s = 0;
    for (int t = 0; t < nk; ++t) {
        const bool nxt = (t + 1 < nk);
        const int  kt  = (t + 1) * K_BLK;

        // ---- issue global loads for tile t+1 (overlap with WMMA below) ----
        v4f ra[4], rb[4];
        if (nxt) {
            #pragma unroll
            for (int i = 0; i < 4; ++i)
                ra[i] = *(const v4f*)(A + (size_t)(row0 + rA[i]) * K + kt + cA[i]);
            #pragma unroll
            for (int i = 0; i < 4; ++i)
                #pragma unroll
                for (int j = 0; j < 4; ++j)
                    rb[i][j] = B[(size_t)(kt + kqB[i] * 4 + j) * N + col0 + nB[i]];
        }
        if (t + 2 < nk)   // L2 prefetch of the tile after next
            __builtin_prefetch(A + (size_t)(row0 + rA[0]) * K + (t + 2) * K_BLK + cA[0], 0, 1);

        // ---- compute on buffer s ----
        const __bf16* pa = lds_a[s];
        const __bf16* pb = lds_b[s];

        v16bf afrag[4], bfrag[2];
        #pragma unroll
        for (int mi = 0; mi < 4; ++mi) {
            const __bf16* ap = &pa[(wm * 64 + mi * 16 + l16) * KPAD + kba];
            const v8bf lo = *(const v8bf*)ap;          // k = kba .. kba+7
            const v8bf hi = *(const v8bf*)(ap + 16);   // k = 16+kba .. 16+kba+7
            #pragma unroll
            for (int e = 0; e < 8; ++e) { afrag[mi][e] = lo[e]; afrag[mi][8 + e] = hi[e]; }
        }
        #pragma unroll
        for (int ni = 0; ni < 2; ++ni) {
            const __bf16* bp = &pb[(wn * 32 + ni * 16 + l16) * KPAD + kba];
            const v8bf lo = *(const v8bf*)bp;          // k = kba .. kba+7
            const v8bf hi = *(const v8bf*)(bp + 16);   // k = 16+kba .. 16+kba+7
            #pragma unroll
            for (int e = 0; e < 8; ++e) { bfrag[ni][e] = lo[e]; bfrag[ni][8 + e] = hi[e]; }
        }

        #pragma unroll
        for (int mi = 0; mi < 4; ++mi)
            #pragma unroll
            for (int ni = 0; ni < 2; ++ni)
                acc[mi][ni] = __builtin_amdgcn_wmma_f32_16x16x32_bf16(
                    false, afrag[mi], false, bfrag[ni],
                    (short)0, acc[mi][ni], false, false);

        // ---- convert + store tile t+1 into buffer s^1 ----
        if (nxt) {
            __bf16* qa = lds_a[s ^ 1];
            __bf16* qb = lds_b[s ^ 1];
            #pragma unroll
            for (int i = 0; i < 4; ++i)
                *(v4bf*)&qa[rA[i] * KPAD + cA[i]] = cvt4(ra[i]);
            #pragma unroll
            for (int i = 0; i < 4; ++i)
                *(v4bf*)&qb[nB[i] * KPAD + kqB[i] * 4] = cvt4(rb[i]);
        }
        __syncthreads();
        s ^= 1;
    }

    // ---- epilogue: C/D f32 layout: VGPR r -> M = r + 8*(lane>=16), N = lane&15 ----
    const int mofs = (lane < 16) ? 0 : 8;
    #pragma unroll
    for (int mi = 0; mi < 4; ++mi) {
        #pragma unroll
        for (int ni = 0; ni < 2; ++ni) {
            const int col = col0 + wn * 32 + ni * 16 + l16;
            const float bv = bias[col];
            #pragma unroll
            for (int r = 0; r < 8; ++r) {
                const int row = row0 + wm * 64 + mi * 16 + mofs + r;
                float v = acc[mi][ni][r] + bv;
                if (relu) v = v > 0.f ? v : 0.f;
                C[(size_t)row * N + col] = v;
            }
        }
    }
}

// One wave (32 lanes) per row: 16 dot products of length 1024, wave32
// shfl-xor reduction, softmax over the 4 class logits on lane 0.
__global__ __launch_bounds__(256) void heads_kernel(
    const float* __restrict__ H,    // [Nrows, 1024]
    const float* __restrict__ Wc,   // [1024, 4]
    const float* __restrict__ bc,   // [4]
    const float* __restrict__ Wr,   // [1024, 12]
    const float* __restrict__ br,   // [12]
    float* __restrict__ logits,     // [Nrows, 4]
    float* __restrict__ box,        // [Nrows, 12]
    int Nrows)
{
    const int lane = threadIdx.x & 31;
    const int wave = threadIdx.x >> 5;
    const int row  = blockIdx.x * 8 + wave;
    if (row >= Nrows) return;

    const float* h = H + (size_t)row * 1024;
    float accc[4]  = {};
    float accr[12] = {};

    for (int k = lane; k < 1024; k += 32) {
        const float hv = h[k];
        const float* wc = Wc + k * 4;
        #pragma unroll
        for (int j = 0; j < 4; ++j) accc[j] += hv * wc[j];
        const float* wr = Wr + k * 12;
        #pragma unroll
        for (int j = 0; j < 12; ++j) accr[j] += hv * wr[j];
    }

    #pragma unroll
    for (int j = 0; j < 4; ++j)
        #pragma unroll
        for (int off = 16; off > 0; off >>= 1)
            accc[j] += __shfl_xor(accc[j], off, 32);
    #pragma unroll
    for (int j = 0; j < 12; ++j)
        #pragma unroll
        for (int off = 16; off > 0; off >>= 1)
            accr[j] += __shfl_xor(accr[j], off, 32);

    if (lane == 0) {
        float v[4];
        float mx = -INFINITY;
        #pragma unroll
        for (int j = 0; j < 4; ++j) { v[j] = accc[j] + bc[j]; mx = fmaxf(mx, v[j]); }
        float sum = 0.f;
        #pragma unroll
        for (int j = 0; j < 4; ++j) { v[j] = __expf(v[j] - mx); sum += v[j]; }
        const float inv = 1.f / sum;
        #pragma unroll
        for (int j = 0; j < 4; ++j) logits[(size_t)row * 4 + j] = v[j] * inv;
        #pragma unroll
        for (int j = 0; j < 12; ++j) box[(size_t)row * 12 + j] = accr[j] + br[j];
    }
}

extern "C" void kernel_launch(void* const* d_in, const int* in_sizes, int n_in,
                              void* d_out, int out_size, void* d_ws, size_t ws_size,
                              hipStream_t stream) {
    (void)n_in; (void)out_size; (void)ws_size;
    const int D_IN = 256 * 7 * 7;   // 12544
    const int HID  = 1024;
    const int Nrows = in_sizes[0] / D_IN;   // 8192

    const float* X  = (const float*)d_in[0];
    const float* W1 = (const float*)d_in[1];
    const float* b1 = (const float*)d_in[2];
    const float* W2 = (const float*)d_in[3];
    const float* b2 = (const float*)d_in[4];
    const float* Wc = (const float*)d_in[5];
    const float* bc = (const float*)d_in[6];
    const float* Wr = (const float*)d_in[7];
    const float* br = (const float*)d_in[8];

    float* H1 = (float*)d_ws;                       // [Nrows, HID] 32 MB
    float* H2 = H1 + (size_t)Nrows * HID;           // [Nrows, HID] 32 MB

    float* logits = (float*)d_out;                  // [Nrows, 4]
    float* box    = logits + (size_t)Nrows * 4;     // [Nrows, 12]

    dim3 grid(HID / N_BLK, Nrows / M_BLK);          // bn fast: W tiles stay hot in L2
    // H1 = X @ W1 + b1
    gemm_bias_kernel<<<grid, 256, 0, stream>>>(X, W1, b1, H1, Nrows, HID, D_IN, 0);
    // H2 = relu(H1 @ W2 + b2)
    gemm_bias_kernel<<<grid, 256, 0, stream>>>(H1, W2, b2, H2, Nrows, HID, HID, 1);
    // heads
    heads_kernel<<<Nrows / 8, 256, 0, stream>>>(H2, Wc, bc, Wr, br, logits, box, Nrows);
}
